// MultiHeadAttention_37477884625313
// MI455X (gfx1250) — matrix operand order repaired and checked by reading, not compiled, not run on hardware
//
#include <hip/hip_runtime.h>

// Relative-position MHA (VITS-style), B=4, C=512, T=1024, H=8, d=64, window=4.
// The padded relative embeddings are non-zero only on a +/-4 band, so rel-logits
// reduce to a banded bias q[t].emb_k[s-t+4] and the rel-v term to a 9-tap banded
// correction on ctx. Pipeline: QKV GEMMs (WMMA f16 in / f32 acc) -> banded flash
// attention -> output GEMM. Workspace ~35MB, L2-resident on MI455X (192MB L2).
// GEMM kernels use 32x32 per-wave register blocking: 4 WMMA per 4 tile loads.

typedef __attribute__((ext_vector_type(16))) _Float16 v16h;
typedef __attribute__((ext_vector_type(8)))  float    v8f;

#define B_N    4
#define CH     512
#define NH     8
#define KCH    64
#define T_LEN  1024
#define NBH    (B_N * NH)   // 32

__device__ __forceinline__ v8f wmma16(v16h a, v16h b, v8f c) {
  return __builtin_amdgcn_wmma_f32_16x16x32_f16(false, a, false, b, (short)0, c,
                                                false, false);
}

// A-matrix tile (16 rows x 32 K). lane&15 = row; lanes 0-15 hold K hi*8+{0..7}
// in elems 0..7 and K hi*8+{16..23} in elems 8..15 (hi = lane>>4).
__device__ __forceinline__ v16h load_a(const float* __restrict__ base,
                                       int row_stride, int k_stride) {
  const int lane = threadIdx.x & 31;
  const float* p = base + (lane & 15) * row_stride + (lane >> 4) * 8 * k_stride;
  v16h o;
#pragma unroll
  for (int i = 0; i < 8; ++i) o[i] = (_Float16)p[i * k_stride];
  p += 16 * k_stride;
#pragma unroll
  for (int i = 0; i < 8; ++i) o[8 + i] = (_Float16)p[i * k_stride];
  return o;
}

__device__ __forceinline__ v16h load_a_lds(const _Float16* __restrict__ base,
                                           int row_stride) {
  const int lane = threadIdx.x & 31;
  const _Float16* p = base + (lane & 15) * row_stride + (lane >> 4) * 8;
  v16h o;
#pragma unroll
  for (int i = 0; i < 8; ++i) o[i] = p[i];
#pragma unroll
  for (int i = 0; i < 8; ++i) o[8 + i] = p[16 + i];
  return o;
}

// B-matrix tile (32 K x 16 cols). lane&15 = column; lanes hold K hi*8+{0..7}
// in elems 0..7 and K 16+hi*8+{0..7} in elems 8..15 (hi = lane>>4), matching A.
__device__ __forceinline__ v16h load_b(const float* __restrict__ base,
                                       int col_stride, int k_stride) {
  const int lane = threadIdx.x & 31;
  const float* p = base + (lane & 15) * col_stride + (lane >> 4) * 8 * k_stride;
  v16h o;
#pragma unroll
  for (int i = 0; i < 8; ++i) o[i] = (_Float16)p[i * k_stride];
  p += 16 * k_stride;
#pragma unroll
  for (int i = 0; i < 8; ++i) o[8 + i] = (_Float16)p[i * k_stride];
  return o;
}

// ---------------- QKV projection: dst[b][h][t][d] = W @ x + bias (q scaled) ---
// Workgroup tile 64(M) x 128(N); 8 waves = 2x4 grid of 32x32 quadrants.
__global__ __launch_bounds__(256) void proj_qkv_kernel(
    const float* __restrict__ x_q, const float* __restrict__ x_k,
    const float* __restrict__ x_v, const float* __restrict__ Wq,
    const float* __restrict__ bq, const float* __restrict__ Wk,
    const float* __restrict__ bk, const float* __restrict__ Wv,
    const float* __restrict__ bv, float* __restrict__ qh,
    float* __restrict__ kh, float* __restrict__ vh) {
  const int z = blockIdx.z;
  const int b = z / 3, p = z % 3;
  const float* x    = (p == 0) ? x_q : (p == 1) ? x_k : x_v;
  const float* W    = (p == 0) ? Wq  : (p == 1) ? Wk  : Wv;
  const float* bias = (p == 0) ? bq  : (p == 1) ? bk  : bv;
  float* dst        = (p == 0) ? qh  : (p == 1) ? kh  : vh;
  const float scl   = (p == 0) ? 0.125f : 1.0f;   // 1/sqrt(64) on q

  const int mb = blockIdx.y * 64;         // output-channel tile
  const int tb = blockIdx.x * 128;        // time tile
  const int w = threadIdx.x >> 5;
  const int lane = threadIdx.x & 31;
  const int r2 = lane & 15, hi = lane >> 4;
  const int mq = w >> 2;                  // 0..1 -> 32-row slab
  const int nq = w & 3;                   // 0..3 -> 32-col slab

  v8f acc00 = {}, acc01 = {}, acc10 = {}, acc11 = {};
  const float* Wbase = W + (size_t)(mb + mq * 32) * CH;
  const float* xcol  = x + (size_t)b * CH * T_LEN + tb + nq * 32;
#pragma unroll 2
  for (int kb = 0; kb < CH; kb += 32) {
    v16h a0 = load_a(Wbase + kb, CH, 1);
    v16h a1 = load_a(Wbase + (size_t)16 * CH + kb, CH, 1);
    v16h b0 = load_b(xcol + (size_t)kb * T_LEN, 1, T_LEN);
    v16h b1 = load_b(xcol + (size_t)kb * T_LEN + 16, 1, T_LEN);
    acc00 = wmma16(a0, b0, acc00);
    acc01 = wmma16(a0, b1, acc01);
    acc10 = wmma16(a1, b0, acc10);
    acc11 = wmma16(a1, b1, acc11);
  }
  const int cb = mb + mq * 32;
  const int tb2 = tb + nq * 32;
#pragma unroll
  for (int r = 0; r < 8; ++r) {
    const int rr = r + hi * 8;                 // D layout: M = r + hi*8
    const int c0 = cb + rr, c1 = cb + 16 + rr;
    const int h0 = c0 >> 6, d0 = c0 & 63;
    const int h1 = c1 >> 6, d1 = c1 & 63;
    const int t0 = tb2 + r2, t1 = tb2 + 16 + r2;  // D layout: N = lane&15
    const float v00 = (acc00[r] + bias[c0]) * scl;
    const float v01 = (acc01[r] + bias[c0]) * scl;
    const float v10 = (acc10[r] + bias[c1]) * scl;
    const float v11 = (acc11[r] + bias[c1]) * scl;
    dst[((size_t)(b * NH + h0) * T_LEN + t0) * KCH + d0] = v00;
    dst[((size_t)(b * NH + h0) * T_LEN + t1) * KCH + d0] = v01;
    dst[((size_t)(b * NH + h1) * T_LEN + t0) * KCH + d1] = v10;
    dst[((size_t)(b * NH + h1) * T_LEN + t1) * KCH + d1] = v11;
  }
}

// ---------------- banded rel-k bias table: bias[bh][t][j] = q[t].emb_k[j] -----
__global__ __launch_bounds__(256) void rel_bias_kernel(
    const float* __restrict__ qh, const float* __restrict__ emb_rel_k,
    float* __restrict__ bias_k) {
  const int gid = blockIdx.x * 256 + threadIdx.x;  // 32768 = NBH * T
  const int bh = gid >> 10;
  const int t = gid & (T_LEN - 1);
  const int h = bh & (NH - 1);
  const float* q = qh + ((size_t)bh * T_LEN + t) * KCH;
#pragma unroll
  for (int j = 0; j < 9; ++j) {
    const float* e = emb_rel_k + (h * 9 + j) * KCH;
    float dot = 0.f;
#pragma unroll
    for (int d = 0; d < KCH; ++d) dot += q[d] * e[d];
    bias_k[((size_t)bh * T_LEN + t) * 9 + j] = dot;
  }
}

// ---------------- banded flash attention per (b,h, 64-query block) -----------
__global__ __launch_bounds__(256) void attention_kernel(
    const float* __restrict__ qh, const float* __restrict__ kh,
    const float* __restrict__ vh, const float* __restrict__ bias_k,
    const float* __restrict__ emb_rel_v, float* __restrict__ ctx) {
  __shared__ float Scorr[64 * 64];       // score tiles 64x32 / corr 64x64
  __shared__ _Float16 P_lds[64 * 32];    // exp-weights for P @ V
  __shared__ float m_lds[64], l_lds[64], sc_lds[64];
  __shared__ float band_p[64 * 9];

  const int bh = blockIdx.y;
  const int h = bh & (NH - 1);
  const int qb0 = blockIdx.x * 64;
  const int tid = threadIdx.x;
  const int w = tid >> 5;
  const int lane = tid & 31;
  const int r2 = lane & 15, hi = lane >> 4;
  const int qt = w >> 1;                 // query sub-tile (score & ctx phases)
  const int kt = w & 1;                  // key sub-tile (score phase)
  const int d0 = (w & 1) * 32;           // 32-wide d slab (ctx phase)

  if (tid < 64) { m_lds[tid] = -1e30f; l_lds[tid] = 0.f; }

  const float* qbase = qh + ((size_t)bh * T_LEN + qb0 + qt * 16) * KCH;
  v16h qa0 = load_a(qbase, KCH, 1);      // Q tile held in registers all loop
  v16h qa1 = load_a(qbase + 32, KCH, 1);
  v8f acc0 = {}; v8f acc1 = {};
  __syncthreads();

  for (int sb = 0; sb < T_LEN; sb += 32) {
    // S = Q K^T (16x16 per wave, K-dim 64 = 2 WMMAs) + banded rel-k bias
    const float* kbase = kh + ((size_t)bh * T_LEN + sb + kt * 16) * KCH;
    v16h kb0 = load_b(kbase, KCH, 1);
    v16h kb1 = load_b(kbase + 32, KCH, 1);
    v8f s = {};
    s = wmma16(qa0, kb0, s);
    s = wmma16(qa1, kb1, s);
    const int scol = sb + kt * 16 + r2;
#pragma unroll
    for (int r = 0; r < 8; ++r) {
      const int ql = qt * 16 + r + hi * 8;
      const int t = qb0 + ql;
      const int delta = scol - t;
      float sv = s[r];
      if (delta >= -4 && delta <= 4)
        sv += bias_k[((size_t)bh * T_LEN + t) * 9 + delta + 4];
      Scorr[ql * 32 + kt * 16 + r2] = sv;
    }
    __syncthreads();
    // online softmax, one thread per query row
    if (tid < 64) {
      const float mo = m_lds[tid];
      float bm = -1e30f;
      for (int i = 0; i < 32; ++i) bm = fmaxf(bm, Scorr[tid * 32 + i]);
      const float mn = fmaxf(mo, bm);
      const float sc = __expf(mo - mn);
      float sum = 0.f;
      for (int i = 0; i < 32; ++i) {
        const float pv = __expf(Scorr[tid * 32 + i] - mn);
        sum += pv;
        P_lds[tid * 32 + i] = (_Float16)pv;
      }
      m_lds[tid] = mn;
      l_lds[tid] = l_lds[tid] * sc + sum;
      sc_lds[tid] = sc;
    }
    __syncthreads();
    // rescale running ctx, accumulate P(16x32) @ V(32x16) x2
    v16h pa = load_a_lds(P_lds + qt * 16 * 32, 32);
    const float* vbase = vh + ((size_t)bh * T_LEN + sb) * KCH + d0;
    v16h vb0 = load_b(vbase, 1, KCH);
    v16h vb1 = load_b(vbase + 16, 1, KCH);
#pragma unroll
    for (int r = 0; r < 8; ++r) {
      const float sc = sc_lds[qt * 16 + r + hi * 8];
      acc0[r] *= sc;
      acc1[r] *= sc;
    }
    acc0 = wmma16(pa, vb0, acc0);
    acc1 = wmma16(pa, vb1, acc1);
    __syncthreads();
  }

  // finalize: 9 band attn probs (recomputed from running max/sum) -> rel-v corr
  if (tid < 64) {
    const int t = qb0 + tid;
    const float mf = m_lds[tid];
    const float inv_l = 1.0f / l_lds[tid];
    sc_lds[tid] = inv_l;
    const float* q = qh + ((size_t)bh * T_LEN + t) * KCH;
#pragma unroll
    for (int j = 0; j < 9; ++j) {
      const int s = t + j - 4;
      float pv = 0.f;
      if (s >= 0 && s < T_LEN) {
        const float* kk = kh + ((size_t)bh * T_LEN + s) * KCH;
        float dot = bias_k[((size_t)bh * T_LEN + t) * 9 + j];
        for (int d = 0; d < KCH; ++d) dot += q[d] * kk[d];
        pv = __expf(dot - mf) * inv_l;
      }
      band_p[tid * 9 + j] = pv;
    }
  }
  __syncthreads();
  {
    const int row = tid >> 2;
    const int dq = (tid & 3) * 16;
#pragma unroll
    for (int dd = 0; dd < 16; ++dd) {
      const int d = dq + dd;
      float c = 0.f;
#pragma unroll
      for (int j = 0; j < 9; ++j)
        c += band_p[row * 9 + j] * emb_rel_v[(h * 9 + j) * KCH + d];
      Scorr[row * 64 + d] = c;
    }
  }
  __syncthreads();
#pragma unroll
  for (int r = 0; r < 8; ++r) {
    const int ql = qt * 16 + r + hi * 8;
    const int t = qb0 + ql;
    const float inv_l = sc_lds[ql];
    const int da = d0 + r2;
    const int db = d0 + 16 + r2;
    float* out = ctx + ((size_t)bh * T_LEN + t) * KCH;
    out[da] = acc0[r] * inv_l + Scorr[ql * 64 + da];
    out[db] = acc1[r] * inv_l + Scorr[ql * 64 + db];
  }
}

// ---------------- output projection: out[b][c][t] = Wo @ ctx + bo ------------
// Workgroup tile 64(M) x 128(N); 8 waves = 2x4 grid of 32x32 quadrants.
__global__ __launch_bounds__(256) void out_proj_kernel(
    const float* __restrict__ ctx, const float* __restrict__ Wo,
    const float* __restrict__ bo, float* __restrict__ out) {
  const int b = blockIdx.z;
  const int mb = blockIdx.y * 64;
  const int tb = blockIdx.x * 128;
  const int w = threadIdx.x >> 5;
  const int lane = threadIdx.x & 31;
  const int r2 = lane & 15, hi = lane >> 4;
  const int mq = w >> 2;                 // 0..1
  const int nq = w & 3;                  // 0..3

  v8f acc00 = {}, acc01 = {}, acc10 = {}, acc11 = {};
  const float* Wbase = Wo + (size_t)(mb + mq * 32) * CH;
#pragma unroll 2
  for (int kb = 0; kb < CH; kb += 32) {
    v16h a0 = load_a(Wbase + kb, CH, 1);
    v16h a1 = load_a(Wbase + (size_t)16 * CH + kb, CH, 1);
    // ctx channel c = h*64+d: a 32-wide K chunk stays inside one head (d-contig)
    const float* bsrc = ctx +
        (((size_t)b * NH + (kb >> 6)) * T_LEN + tb + nq * 32) * KCH + (kb & 63);
    v16h b0 = load_b(bsrc, KCH, 1);
    v16h b1 = load_b(bsrc + (size_t)16 * KCH, KCH, 1);
    acc00 = wmma16(a0, b0, acc00);
    acc01 = wmma16(a0, b1, acc01);
    acc10 = wmma16(a1, b0, acc10);
    acc11 = wmma16(a1, b1, acc11);
  }
  const int cb = mb + mq * 32;
  const int tb2 = tb + nq * 32;
#pragma unroll
  for (int r = 0; r < 8; ++r) {
    const int rr = r + hi * 8;
    const int c0 = cb + rr, c1 = cb + 16 + rr;
    const int t0 = tb2 + r2, t1 = tb2 + 16 + r2;
    out[((size_t)b * CH + c0) * T_LEN + t0]      = acc00[r] + bo[c0];
    out[((size_t)b * CH + c0) * T_LEN + t1]      = acc01[r] + bo[c0];
    out[((size_t)b * CH + c1) * T_LEN + t0]      = acc10[r] + bo[c1];
    out[((size_t)b * CH + c1) * T_LEN + t1]      = acc11[r] + bo[c1];
  }
}

extern "C" void kernel_launch(void* const* d_in, const int* in_sizes, int n_in,
                              void* d_out, int out_size, void* d_ws,
                              size_t ws_size, hipStream_t stream) {
  (void)in_sizes; (void)n_in; (void)out_size; (void)ws_size;
  const float* x_q = (const float*)d_in[0];
  const float* x_k = (const float*)d_in[1];
  const float* x_v = (const float*)d_in[2];
  const float* Wq  = (const float*)d_in[3];
  const float* bq  = (const float*)d_in[4];
  const float* Wk  = (const float*)d_in[5];
  const float* bk  = (const float*)d_in[6];
  const float* Wv  = (const float*)d_in[7];
  const float* bv  = (const float*)d_in[8];
  const float* Wo  = (const float*)d_in[9];
  const float* bo  = (const float*)d_in[10];
  const float* emb_rel_k = (const float*)d_in[11];
  const float* emb_rel_v = (const float*)d_in[12];
  float* out = (float*)d_out;

  // workspace layout (floats): qh | kh | vh | ctx | bias_k  (~34.7 MB total)
  float* ws = (float*)d_ws;
  const size_t HEAD = (size_t)NBH * T_LEN * KCH;  // 2,097,152 floats each
  float* qh = ws;
  float* kh = qh + HEAD;
  float* vh = kh + HEAD;
  float* ctxb = vh + HEAD;
  float* bias_k = ctxb + HEAD;                    // NBH*T*9 floats

  proj_qkv_kernel<<<dim3(8, 8, 12), 256, 0, stream>>>(
      x_q, x_k, x_v, Wq, bq, Wk, bk, Wv, bv, qh, kh, vh);
  rel_bias_kernel<<<dim3(128), 256, 0, stream>>>(qh, emb_rel_k, bias_k);
  attention_kernel<<<dim3(16, 32), 256, 0, stream>>>(qh, kh, vh, bias_k,
                                                     emb_rel_v, ctxb);
  out_proj_kernel<<<dim3(8, 8, 4), 256, 0, stream>>>(ctxb, Wo, bo, out);
}
